// FeatureSimilarity_73667279061041
// MI455X (gfx1250) — matrix operand order repaired and checked by reading, not compiled, not run on hardware
//
#include <hip/hip_runtime.h>

// ---------------------------------------------------------------------------
// FeatureSimilarity on MI455X (gfx1250): -||f_i - f_j||_2 via split-bf16 WMMA
//   gram ~= Hi*Hi^T + Hi*Lo^T + Lo*Hi^T  (f32 accumulate, 16x16x32 bf16 WMMA)
//   out  = -sqrt(max(sq[i] + sq[j] - 2*gram, 0))
// Data movement: double-buffered GLOBAL_LOAD_ASYNC_TO_LDS_B128 (ASYNCcnt).
// ---------------------------------------------------------------------------

#define NROW 8192
#define DDIM 512
#define KC   32            // K-chunk per LDS stage (one WMMA K-step)
#define BM   128           // block tile rows
#define BN   128           // block tile cols
#define LDSP (KC + 8)      // padded row stride in ushorts: 40*2B = 80B (16B-aligned)
#define PANEL (128 * LDSP) // one matrix component panel, in ushorts
#define STAGE (4 * PANEL)  // Ahi | Alo | Bhi | Blo

typedef __attribute__((ext_vector_type(16))) __bf16 v16bf;
typedef __attribute__((ext_vector_type(8)))  float  v8f;

union Frag {               // 16 bf16 = 32 bytes = 8 VGPRs (WMMA A/B operand)
    v16bf v;
    uint4 u[2];
};

__device__ __forceinline__ unsigned short f32_to_bf16_rne(float f) {
    unsigned int x = __float_as_uint(f);
    unsigned int lsb = (x >> 16) & 1u;
    x += 0x7fffu + lsb;                    // round to nearest even
    return (unsigned short)(x >> 16);
}

// gfx1250 async memory->LDS copy, tracked with ASYNCcnt (ISA 15.18.3 op 98).
__device__ __forceinline__ void async_b128(unsigned lds_byte_off, const void* gptr) {
    asm volatile("global_load_async_to_lds_b128 %0, %1, off"
                 :: "v"(lds_byte_off), "v"(gptr) : "memory");
}
__device__ __forceinline__ void wait_async_le8() {
    asm volatile("s_wait_asynccnt 0x8" ::: "memory");
}
__device__ __forceinline__ void wait_async_0() {
    asm volatile("s_wait_asynccnt 0x0" ::: "memory");
}

// ---------------------------------------------------------------------------
// Prep: per-row squared norm + split each f32 into bf16 hi/lo planes.
// One wave32 per row; 8 rows per 256-thread block.
// ---------------------------------------------------------------------------
__global__ __launch_bounds__(256)
void FeatureSimilarity_prep_kernel(const float* __restrict__ f,
                                   unsigned short* __restrict__ fhi,
                                   unsigned short* __restrict__ flo,
                                   float* __restrict__ sq)
{
    const int wave = threadIdx.x >> 5;
    const int lane = threadIdx.x & 31;
    const int row  = blockIdx.x * 8 + wave;

    const float* frow = f + (size_t)row * DDIM;
    float ss = 0.0f;

    #pragma unroll
    for (int c = 0; c < 4; ++c) {
        const int col = c * 128 + lane * 4;          // coalesced float4
        float4 v = *(const float4*)(frow + col);
        float vv[4] = {v.x, v.y, v.z, v.w};
        unsigned short h[4], l[4];
        #pragma unroll
        for (int j = 0; j < 4; ++j) {
            ss = fmaf(vv[j], vv[j], ss);
            unsigned short hb = f32_to_bf16_rne(vv[j]);
            float hf = __uint_as_float(((unsigned int)hb) << 16);
            h[j] = hb;
            l[j] = f32_to_bf16_rne(vv[j] - hf);
        }
        unsigned hp0 = (unsigned)h[0] | ((unsigned)h[1] << 16);
        unsigned hp1 = (unsigned)h[2] | ((unsigned)h[3] << 16);
        unsigned lp0 = (unsigned)l[0] | ((unsigned)l[1] << 16);
        unsigned lp1 = (unsigned)l[2] | ((unsigned)l[3] << 16);
        *(uint2*)(fhi + (size_t)row * DDIM + col) = make_uint2(hp0, hp1);
        *(uint2*)(flo + (size_t)row * DDIM + col) = make_uint2(lp0, lp1);
    }

    #pragma unroll
    for (int off = 16; off > 0; off >>= 1)
        ss += __shfl_down(ss, off, 32);
    if (lane == 0) sq[row] = ss;
}

// ---------------------------------------------------------------------------
// Main GEMM + distance epilogue.
// Block: 256 threads (8 waves) -> 128x128 output tile.
// Waves: 4 (M) x 2 (N); each wave owns 32x64 = 2x4 WMMA 16x16 tiles.
// Double-buffered async LDS staging: 2 stages x (Ahi|Alo|Bhi|Blo) x 128x40.
// ---------------------------------------------------------------------------
__global__ __launch_bounds__(256)
void FeatureSimilarity_wmma_kernel(const unsigned short* __restrict__ fhi,
                                   const unsigned short* __restrict__ flo,
                                   const float* __restrict__ sq,
                                   float* __restrict__ out)
{
    __shared__ __align__(16) unsigned short lds[2 * STAGE];   // 80 KB

    // Escape the LDS array with provenance: the async copies below write it
    // from inline asm, so the compiler must not assume it is never stored to.
    asm volatile("" :: "s"((void*)lds) : "memory");

    const int tid   = threadIdx.x;
    const int lane  = tid & 31;
    const int wave  = tid >> 5;
    const int waveM = wave & 3;    // 0..3 -> 32-row strip
    const int waveN = wave >> 2;   // 0..1 -> 64-col strip
    const int rowBase = blockIdx.x * BM;
    const int colBase = blockIdx.y * BN;

    v8f acc[2][4] = {};

    // Per-thread async slots (b128 = 8 ushorts):
    //   A panel 128x32 ushorts = 512 b128 -> rows tid>>2 and tid>>2+64
    //   B panel 128x32 ushorts = 512 b128 -> same rows
    const int prow0 = tid >> 2;
    const int prow1 = prow0 + 64;
    const int pcc   = (tid & 3) * 8;                  // ushort col within K-chunk

    // issue all 8 async b128 copies for one stage / one K-chunk
    auto issueStage = [&](int stage, int kb) {
        const unsigned b0 = (unsigned)(stage * STAGE) * 2u;   // stage base, bytes
        const unsigned la0 = b0 + (unsigned)(prow0 * LDSP + pcc) * 2u;
        const unsigned la1 = b0 + (unsigned)(prow1 * LDSP + pcc) * 2u;
        const size_t  gA0 = (size_t)(rowBase + prow0) * DDIM + kb + pcc;
        const size_t  gA1 = (size_t)(rowBase + prow1) * DDIM + kb + pcc;
        const size_t  gB0 = (size_t)(colBase + prow0) * DDIM + kb + pcc;
        const size_t  gB1 = (size_t)(colBase + prow1) * DDIM + kb + pcc;
        async_b128(la0,                 fhi + gA0);           // Ahi
        async_b128(la1,                 fhi + gA1);
        async_b128(la0 + PANEL * 2,     flo + gA0);           // Alo
        async_b128(la1 + PANEL * 2,     flo + gA1);
        async_b128(la0 + 2 * PANEL * 2, fhi + gB0);           // Bhi
        async_b128(la1 + 2 * PANEL * 2, fhi + gB1);
        async_b128(la0 + 3 * PANEL * 2, flo + gB0);           // Blo
        async_b128(la1 + 3 * PANEL * 2, flo + gB1);
    };

    // consume one staged K-chunk: 24 WMMAs from LDS fragments
    auto computeChunk = [&](int stage) {
        const unsigned short* L = lds + stage * STAGE;
        const int half = lane >> 4;      // 0: lanes 0-15, 1: lanes 16-31
        const int mn   = lane & 15;
        const int ka   = half * 8;       // A: K chunks {ka..ka+7} and {ka+16..ka+23}
        const int kbb  = half * 8;       // B: K chunks {kbb..kbb+7} and {kbb+16..kbb+23}

        Frag ahi[2], alo[2], bhi[4], blo[4];
        #pragma unroll
        for (int t = 0; t < 2; ++t) {
            const int m = waveM * 32 + t * 16 + mn;
            ahi[t].u[0] = *(const uint4*)&L[0 * PANEL + m * LDSP + ka];
            ahi[t].u[1] = *(const uint4*)&L[0 * PANEL + m * LDSP + ka + 16];
            alo[t].u[0] = *(const uint4*)&L[1 * PANEL + m * LDSP + ka];
            alo[t].u[1] = *(const uint4*)&L[1 * PANEL + m * LDSP + ka + 16];
        }
        #pragma unroll
        for (int u = 0; u < 4; ++u) {
            const int n = waveN * 64 + u * 16 + mn;
            bhi[u].u[0] = *(const uint4*)&L[2 * PANEL + n * LDSP + kbb];
            bhi[u].u[1] = *(const uint4*)&L[2 * PANEL + n * LDSP + kbb + 16];
            blo[u].u[0] = *(const uint4*)&L[3 * PANEL + n * LDSP + kbb];
            blo[u].u[1] = *(const uint4*)&L[3 * PANEL + n * LDSP + kbb + 16];
        }

        // split-precision accumulation: HiHi + HiLo + LoHi
        #pragma unroll
        for (int t = 0; t < 2; ++t) {
            #pragma unroll
            for (int u = 0; u < 4; ++u) {
                acc[t][u] = __builtin_amdgcn_wmma_f32_16x16x32_bf16(
                    false, ahi[t].v, false, bhi[u].v, (short)0, acc[t][u], false, false);
                acc[t][u] = __builtin_amdgcn_wmma_f32_16x16x32_bf16(
                    false, ahi[t].v, false, blo[u].v, (short)0, acc[t][u], false, false);
                acc[t][u] = __builtin_amdgcn_wmma_f32_16x16x32_bf16(
                    false, alo[t].v, false, bhi[u].v, (short)0, acc[t][u], false, false);
            }
        }
    };

    issueStage(0, 0);

    int s = 0;
    for (int kb = 0; kb < DDIM - KC; kb += KC, s ^= 1) {
        issueStage(s ^ 1, kb + KC);   // overlap next chunk with this chunk's compute
        wait_async_le8();             // in-order: drains my stage-s copies only
        __syncthreads();              // everyone's stage-s copies visible
        computeChunk(s);
        __syncthreads();              // everyone done reading stage s before overwrite
    }
    wait_async_0();                   // last stage
    __syncthreads();
    computeChunk(s);

    // --- epilogue: d2 = sq[i] + sq[j] - 2*gram; out = -sqrt(max(d2,0)) ---
    // C layout: VGPR r, lanes 0-15 -> (M=r,   N=lane); lanes 16-31 -> (M=r+8, N=lane-16)
    const int half = lane >> 4;
    #pragma unroll
    for (int t = 0; t < 2; ++t) {
        const int gmBase = rowBase + waveM * 32 + t * 16 + half * 8;
        #pragma unroll
        for (int u = 0; u < 4; ++u) {
            const int gc = colBase + waveN * 64 + u * 16 + (lane & 15);
            const float sqc = sq[gc];
            #pragma unroll
            for (int r = 0; r < 8; ++r) {
                const int gm = gmBase + r;
                float d2 = sq[gm] + sqc - 2.0f * acc[t][u][r];
                d2 = fmaxf(d2, 0.0f);
                out[(size_t)gm * NROW + gc] = -sqrtf(d2);
            }
        }
    }
}

// ---------------------------------------------------------------------------
extern "C" void kernel_launch(void* const* d_in, const int* in_sizes, int n_in,
                              void* d_out, int out_size, void* d_ws, size_t ws_size,
                              hipStream_t stream)
{
    const float* features = (const float*)d_in[0];
    float* out = (float*)d_out;

    // workspace: [Fhi 8MB][Flo 8MB][sq 32KB]
    unsigned short* fhi = (unsigned short*)d_ws;
    unsigned short* flo = fhi + (size_t)NROW * DDIM;
    float*          sq  = (float*)(flo + (size_t)NROW * DDIM);

    FeatureSimilarity_prep_kernel<<<NROW / 8, 256, 0, stream>>>(features, fhi, flo, sq);

    dim3 grid(NROW / BM, NROW / BN);
    FeatureSimilarity_wmma_kernel<<<grid, 256, 0, stream>>>(fhi, flo, sq, out);
}
